// ModulatedConv2D_9998683865242
// MI455X (gfx1250) — matrix operand order repaired and checked by reading, not compiled, not run on hardware
//
#include <hip/hip_runtime.h>
#include <hip/hip_bf16.h>

#define Bsz  8
#define IC   512
#define OC   512
#define Hdim 32
#define Wdim 32

// fan-in scalings
#define RC_DENSE 0.04419417382415922f      // 1/sqrt(512)
#define RC_CONV  0.01473139127471974f      // 1/sqrt(9*512)

// LDS tile geometry: 3 rows x (32+2 halo) cols x 32 ic, ic-stride padded to 40
#define LW   34
#define STR  40

typedef __attribute__((ext_vector_type(16))) __bf16 v16bf;
typedef __attribute__((ext_vector_type(8)))  float  v8f;

union V16 { v16bf v; uint4 u[2]; };

// ---------------------------------------------------------------------------
// 1) style[b,ic] = RC_DENSE * (w @ dense_w)[b,ic] + dense_b[ic] + 1
// ---------------------------------------------------------------------------
__global__ void style_kernel(const float* __restrict__ w,
                             const float* __restrict__ dw,
                             const float* __restrict__ db,
                             float* __restrict__ style,
                             float* __restrict__ style2) {
    const int b = blockIdx.x;
    const int t = threadIdx.x;            // 256 threads -> 2 ic each
    const float* wb = w + b * IC;
    float a0 = 0.f, a1 = 0.f;
    for (int k = 0; k < IC; ++k) {
        float wv = wb[k];
        a0 += wv * dw[k * IC + t];
        a1 += wv * dw[k * IC + t + 256];
    }
    float s0 = RC_DENSE * a0 + db[t]       + 1.0f;
    float s1 = RC_DENSE * a1 + db[t + 256] + 1.0f;
    style [b * IC + t]        = s0;
    style2[b * IC + t]        = s0 * s0;
    style [b * IC + t + 256]  = s1;
    style2[b * IC + t + 256]  = s1 * s1;
}

// ---------------------------------------------------------------------------
// 2) d[b,oc] = rsqrt(RC_CONV^2 * sum_ic style2[b,ic]*sum_tap conv_w^2 + 1e-8)
// ---------------------------------------------------------------------------
__global__ void demod_kernel(const float* __restrict__ cw,
                             const float* __restrict__ style2,
                             float* __restrict__ dmod) {
    const int oc = blockIdx.x * 256 + threadIdx.x;
    float acc[Bsz];
#pragma unroll
    for (int b = 0; b < Bsz; ++b) acc[b] = 0.f;
    for (int ic = 0; ic < IC; ++ic) {
        float s = 0.f;
#pragma unroll
        for (int tap = 0; tap < 9; ++tap) {
            float v = cw[(tap * IC + ic) * OC + oc];
            s += v * v;
        }
#pragma unroll
        for (int b = 0; b < Bsz; ++b) acc[b] += s * style2[b * IC + ic];
    }
    const float rc2 = RC_CONV * RC_CONV;
#pragma unroll
    for (int b = 0; b < Bsz; ++b)
        dmod[b * OC + oc] = rsqrtf(rc2 * acc[b] + 1e-8f);
}

// ---------------------------------------------------------------------------
// 3) wbf[tap][oc][ic] = bf16(RC_CONV * conv_w[tap][ic][oc])   (A-operand layout)
// ---------------------------------------------------------------------------
__global__ void wprep_kernel(const float* __restrict__ cw,
                             __bf16* __restrict__ wbf) {
    const int oc = blockIdx.x, tap = blockIdx.y, t = threadIdx.x;
#pragma unroll
    for (int i = 0; i < 2; ++i) {
        int ic = t + i * 256;
        wbf[(tap * OC + oc) * IC + ic] =
            (__bf16)(RC_CONV * cw[(tap * IC + ic) * OC + oc]);
    }
}

// ---------------------------------------------------------------------------
// 4) xm[b][h][w][ic] = bf16(x[b][ic][h][w] * style[b][ic])    (NHWC, B-operand)
// ---------------------------------------------------------------------------
__global__ void xmod_kernel(const float* __restrict__ x,
                            const float* __restrict__ style,
                            __bf16* __restrict__ xm) {
    const int h = blockIdx.x, b = blockIdx.y, t = threadIdx.x;
#pragma unroll 4
    for (int iter = 0; iter < (Wdim * IC) / 256; ++iter) {
        int idx = iter * 256 + t;           // linear over (w, ic)
        int ic  = idx & (IC - 1);
        int w   = idx >> 9;
        float v = x[((b * IC + ic) * Hdim + h) * Wdim + w] * style[b * IC + ic];
        xm[((b * Hdim + h) * Wdim + w) * IC + ic] = (__bf16)v;
    }
}

// ---------------------------------------------------------------------------
// 5) implicit-GEMM 3x3 conv with bf16 WMMA, demod on epilogue
//    block = 256 thr (8 waves): 128 oc x 32 pixels (one image row)
//    each wave: two 16x16 tiles (32 oc x 16 pix), shared B operand
//    grid  = (H, OC/128, B); LDS double-buffered + halo-padded (no branches
//    in the tap loop), one barrier per ic-block, operand fetch software-
//    pipelined one tap ahead so WMMA overlaps L0/L2 loads within a wave
// ---------------------------------------------------------------------------
__global__ void __launch_bounds__(256)
conv_kernel(const __bf16* __restrict__ xm,
            const __bf16* __restrict__ wbf,
            const float*  __restrict__ dmod,
            float*        __restrict__ y) {
    __shared__ __align__(16) __bf16 stile[2][3 * LW * STR];

    const int h     = blockIdx.x;
    const int ocblk = blockIdx.y;          // 0..3 -> 128 oc
    const int b     = blockIdx.z;
    const int t     = threadIdx.x;
    const int wave  = t >> 5, lane = t & 31;
    const int oc32  = wave & 3;            // 32-oc group within block
    const int w0    = (wave >> 2) * 16;    // 16-pixel group within row
    const int ocbase = ocblk * 128 + oc32 * 32;

    const int arow = lane & 15;            // A: oc row
    const int akb  = (lane >> 4) * 8;      // A: k base (second half at +16)
    const int bn   = lane & 15;            // B: pixel column
    const int bks  = (lane >> 4) * 8;      // B: k base (second half at +16)

    // zero both buffers once (covers w-halo columns + out-of-range h rows)
    {
        uint4 zero4 = make_uint4(0u, 0u, 0u, 0u);
        uint4* z = (uint4*)&stile[0][0];
        const int n16 = 2 * 3 * LW * STR * 2 / 16;   // 1020 uint4
        for (int i = t; i < n16; i += 256) z[i] = zero4;
    }

    // staging: rows h-1..h+1, 32 w, 32 ic of NHWC bf16 -> halo-padded LDS
    auto stage = [&](int icblk, int buf) {
        if (t < 192) {
            int slice = t >> 1;            // 0..95 = (r, w)
            int half  = t & 1;
            int r  = slice >> 5;           // 0..2
            int ww = slice & 31;
            int hh = h - 1 + r;
            if (hh >= 0 && hh < Hdim) {
                const __bf16* src = xm + ((b * Hdim + hh) * Wdim + ww) * IC
                                       + icblk * 32 + half * 16;
                uint4 val = *(const uint4*)src;
                *(uint4*)(&stile[buf][(r * LW + ww + 1) * STR + half * 16]) = val;
                __builtin_prefetch(src + 64, 0, 1);   // speculative: 2 ic-blocks ahead
            }
        }
    };

    // operand fetch for one tap (A from global/L2, B from LDS)
    auto load_tap = [&](int icblk, int tap, const __bf16* sb,
                        V16& a0, V16& a1, V16& bb) {
        const int dh = tap / 3, dw = tap % 3;         // 0..2 (halo absorbs -1)
        const __bf16* ap = wbf + ((tap * OC + ocbase + arow) * IC)
                               + icblk * 32 + akb;
        a0.u[0] = *(const uint4*)(ap);
        a0.u[1] = *(const uint4*)(ap + 16);
        a1.u[0] = *(const uint4*)(ap + 16 * IC);
        a1.u[1] = *(const uint4*)(ap + 16 * IC + 16);
        const __bf16* bp = sb + (dh * LW + w0 + bn + dw) * STR + bks;
        bb.u[0] = *(const uint4*)(bp);
        bb.u[1] = *(const uint4*)(bp + 16);
    };

    v8f acc0 = {}, acc1 = {};

    __syncthreads();            // zero-fill visible to all
    stage(0, 0);
    __syncthreads();

    for (int icblk = 0; icblk < IC / 32; ++icblk) {
        if (icblk + 1 < IC / 32) stage(icblk + 1, (icblk + 1) & 1);
        const __bf16* sb = &stile[icblk & 1][0];

        V16 a0, a1, bb;
        load_tap(icblk, 0, sb, a0, a1, bb);

#pragma unroll
        for (int tap = 0; tap < 9; ++tap) {
            // prefetch next tap's operands before consuming current ones
            V16 na0, na1, nbb;
            if (tap + 1 < 9) load_tap(icblk, tap + 1, sb, na0, na1, nbb);

            acc0 = __builtin_amdgcn_wmma_f32_16x16x32_bf16(
                       false, a0.v, false, bb.v, (short)0, acc0, false, false);
            acc1 = __builtin_amdgcn_wmma_f32_16x16x32_bf16(
                       false, a1.v, false, bb.v, (short)0, acc1, false, false);

            if (tap + 1 < 9) { a0 = na0; a1 = na1; bb = nbb; }
        }
        __syncthreads();
    }

    // epilogue: demodulate and store (C/D layout: m = (lane/16)*8 + r, n = lane%16)
    const int mrow = (lane >> 4) * 8;
    const int wout = w0 + (lane & 15);
    const float* dp0 = dmod + b * OC + ocbase + mrow;
    const float* dp1 = dp0 + 16;
#pragma unroll
    for (int r = 0; r < 8; ++r) {
        int oc0 = ocbase + mrow + r;
        y[((b * OC + oc0) * Hdim + h) * Wdim + wout]        = acc0[r] * dp0[r];
        y[((b * OC + oc0 + 16) * Hdim + h) * Wdim + wout]   = acc1[r] * dp1[r];
    }
}

// ---------------------------------------------------------------------------
extern "C" void kernel_launch(void* const* d_in, const int* in_sizes, int n_in,
                              void* d_out, int out_size, void* d_ws, size_t ws_size,
                              hipStream_t stream) {
    const float* x  = (const float*)d_in[0];   // [8,512,32,32]
    const float* w  = (const float*)d_in[1];   // [8,512]
    const float* cw = (const float*)d_in[2];   // [3,3,512,512]
    const float* dw = (const float*)d_in[3];   // [512,512]
    const float* db = (const float*)d_in[4];   // [512]
    float* y = (float*)d_out;                  // [8,512,32,32]

    char* ws = (char*)d_ws;
    float*  style  = (float*)(ws);                         // 16 KB
    float*  style2 = (float*)(ws + 16384);                 // 16 KB
    float*  dmod   = (float*)(ws + 32768);                 // 16 KB
    __bf16* wbf    = (__bf16*)(ws + 49152);                // 9*512*512*2 = 4.5 MB
    __bf16* xm     = (__bf16*)(ws + 49152 + 9*512*512*2);  // 8*32*32*512*2 = 8 MB

    style_kernel<<<Bsz, 256, 0, stream>>>(w, dw, db, style, style2);
    demod_kernel<<<OC / 256, 256, 0, stream>>>(cw, style2, dmod);
    wprep_kernel<<<dim3(OC, 9), 256, 0, stream>>>(cw, wbf);
    xmod_kernel<<<dim3(Hdim, Bsz), 256, 0, stream>>>(x, style, xm);
    conv_kernel<<<dim3(Hdim, OC / 128, Bsz), 256, 0, stream>>>(xm, wbf, dmod, y);
}
